// Attention_50190987821304
// MI455X (gfx1250) — matrix operand order repaired and checked by reading, not compiled, not run on hardware
//
#include <hip/hip_runtime.h>
#include <math.h>

typedef __attribute__((ext_vector_type(16))) _Float16 v16h;
typedef __attribute__((ext_vector_type(8)))  float    v8f;

#define SQ   4096
#define DK   64
#define NB   4
#define KBLK 64            // keys staged per iteration
#define WG_Q 64            // query rows per workgroup
#define NWAVE (WG_Q / 16)  // 4 waves
#define NTHREADS (NWAVE * 32)

struct alignas(8) H4 { _Float16 x, y, z, w; };

__device__ __forceinline__ v16h lds_frag(const _Float16* src) {
    v16h f;
    ((uint4*)&f)[0] = *(const uint4*)(src);
    ((uint4*)&f)[1] = *(const uint4*)(src + 8);
    return f;
}

__global__ __launch_bounds__(NTHREADS)
void fa_fwd_f16wmma(const float* __restrict__ Q,
                    const float* __restrict__ K,
                    const float* __restrict__ V,
                    float* __restrict__ O)
{
    __shared__ __align__(16) _Float16 k_sm[KBLK * DK];          // [key][d]
    __shared__ __align__(16) _Float16 vt_sm[DK * KBLK];         // [d][key] (transposed V)
    __shared__ __align__(16) _Float16 p_sm[NWAVE * 16 * KBLK];  // per-wave P tile [row][key]

    const int t    = threadIdx.x;
    const int lane = t & 31;
    const int w    = t >> 5;
    const int half = lane >> 4;   // 0 or 1 (wave32 half)
    const int l16  = lane & 15;

    const int tileq = blockIdx.x & ((SQ / WG_Q) - 1);
    const int b     = blockIdx.x >> 6;   // SQ/WG_Q == 64 tiles per batch
    const int qbase = tileq * WG_Q;
    const int q0    = qbase + w * 16;    // this wave's 16 query rows

    const size_t bofs = (size_t)b * SQ * DK;
    const float* Qb = Q + bofs;
    const float* Kb = K + bofs;
    const float* Vb = V + bofs;
    float*       Ob = O + bofs;

    // base-2 softmax scale folded into Q: logit2 = (q.k) * (1/sqrt(64)) * log2(e)
    const float scale2 = 0.125f * 1.44269504088896f;

    // ---- Q: load 16 rows x 64 d, scale, f32 -> f16, into A-fragment layout.
    // A 16x32 f16: lane holds row (lane&15); half index j maps to
    // k = (lane<16?0:16) + j  -> 16 contiguous d.
    v16h q_a[2];
    {
        const float* qr = Qb + (size_t)(q0 + l16) * DK;
        #pragma unroll
        for (int c = 0; c < 2; ++c) {
            const int d0 = c * 32 + half * 16;
            v16h a;
            #pragma unroll
            for (int j = 0; j < 8; ++j) a[j]     = (_Float16)(qr[d0 + j]      * scale2);
            #pragma unroll
            for (int j = 0; j < 8; ++j) a[8 + j] = (_Float16)(qr[d0 + 8 + j] * scale2);
            q_a[c] = a;
        }
    }

    // Flash-attention state (fp32): O accumulator 16x64 (4 C tiles), row max, row sum.
    v8f o_acc[4];
    #pragma unroll
    for (int dt = 0; dt < 4; ++dt)
        #pragma unroll
        for (int r = 0; r < 8; ++r) o_acc[dt][r] = 0.0f;
    float m_r[8], l_r[8];
    #pragma unroll
    for (int r = 0; r < 8; ++r) { m_r[r] = -3.0e38f; l_r[r] = 0.0f; }

    const int n_kb = qbase / KBLK + 1;    // causal: keys <= qbase+63

    for (int kb = 0; kb < n_kb; ++kb) {
        const int  kbase = kb * KBLK;
        const bool diag  = (kb == n_kb - 1);   // wave-uniform: only last block masks
        const float* Ks = Kb + (size_t)kbase * DK;
        const float* Vs = Vb + (size_t)kbase * DK;

        // ---- stage K [key][d] and V transposed [d][key] as f16 in LDS ----
        #pragma unroll
        for (int p = 0; p < (KBLK * DK) / (NTHREADS * 4); ++p) {
            const int flat = (p * NTHREADS + t) * 4;
            const int key = flat >> 6, d = flat & 63;
            const float4 kf = *(const float4*)(Ks + flat);
            const float4 vf = *(const float4*)(Vs + flat);
            *(H4*)&k_sm[key * DK + d] =
                H4{(_Float16)kf.x, (_Float16)kf.y, (_Float16)kf.z, (_Float16)kf.w};
            vt_sm[(d + 0) * KBLK + key] = (_Float16)vf.x;
            vt_sm[(d + 1) * KBLK + key] = (_Float16)vf.y;
            vt_sm[(d + 2) * KBLK + key] = (_Float16)vf.z;
            vt_sm[(d + 3) * KBLK + key] = (_Float16)vf.w;
            if (!diag) {                      // prefetch next K/V block -> global_prefetch_b8
                __builtin_prefetch(Ks + KBLK * DK + flat, 0, 0);
                __builtin_prefetch(Vs + KBLK * DK + flat, 0, 0);
            }
        }
        __syncthreads();

        // ---- S = Q K^T : 4 key subtiles x (K=64 via 2 WMMAs each) ----
        // B 32x16: lane holds column n=(lane&15)=key; 16 contiguous d at ch*32+half*16.
        float s[4][8];
        #pragma unroll
        for (int nt = 0; nt < 4; ++nt) {
            const _Float16* kb0 = &k_sm[(nt * 16 + l16) * DK + half * 16];
            v16h bf0 = lds_frag(kb0);
            v16h bf1 = lds_frag(kb0 + 32);
            v8f c;
            #pragma unroll
            for (int r = 0; r < 8; ++r) c[r] = 0.0f;
            c = __builtin_amdgcn_wmma_f32_16x16x32_f16(
                    false, q_a[0], false, bf0, (short)0, c, false, false);
            c = __builtin_amdgcn_wmma_f32_16x16x32_f16(
                    false, q_a[1], false, bf1, (short)0, c, false, false);
            #pragma unroll
            for (int r = 0; r < 8; ++r) s[nt][r] = c[r];
        }

        // ---- row max; causal mask only on the diagonal block ----
        const int rowbase = q0 + half * 8;
        float rmax[8];
        if (diag) {
            #pragma unroll
            for (int r = 0; r < 8; ++r) {
                float mx = -3.0e38f;
                const int row = rowbase + r;
                #pragma unroll
                for (int nt = 0; nt < 4; ++nt) {
                    const int key = kbase + nt * 16 + l16;
                    const float v = (key <= row) ? s[nt][r] : -1.0e30f;
                    s[nt][r] = v;
                    mx = fmaxf(mx, v);
                }
                rmax[r] = mx;
            }
        } else {
            #pragma unroll
            for (int r = 0; r < 8; ++r) {
                float mx = fmaxf(fmaxf(s[0][r], s[1][r]), fmaxf(s[2][r], s[3][r]));
                rmax[r] = mx;
            }
        }
        #pragma unroll
        for (int mk = 1; mk < 16; mk <<= 1)
            #pragma unroll
            for (int r = 0; r < 8; ++r)
                rmax[r] = fmaxf(rmax[r], __shfl_xor(rmax[r], mk, 32));

        // ---- online softmax update (exp2; logits already in log2 domain) ----
        float alpha[8], rsum[8];
        #pragma unroll
        for (int r = 0; r < 8; ++r) {
            const float mn = fmaxf(m_r[r], rmax[r]);
            alpha[r] = exp2f(m_r[r] - mn);
            m_r[r]   = mn;
            float acc = 0.0f;
            #pragma unroll
            for (int nt = 0; nt < 4; ++nt) {
                const float p = exp2f(s[nt][r] - mn);
                s[nt][r] = p;
                acc += p;
            }
            rsum[r] = acc;
        }
        #pragma unroll
        for (int mk = 1; mk < 16; mk <<= 1)
            #pragma unroll
            for (int r = 0; r < 8; ++r)
                rsum[r] += __shfl_xor(rsum[r], mk, 32);
        #pragma unroll
        for (int r = 0; r < 8; ++r) {
            l_r[r] = l_r[r] * alpha[r] + rsum[r];
            #pragma unroll
            for (int dt = 0; dt < 4; ++dt) o_acc[dt][r] *= alpha[r];
        }

        // ---- P (C layout) -> per-wave LDS -> A-fragment layout (f16) ----
        _Float16* pw = &p_sm[w * 16 * KBLK];
        #pragma unroll
        for (int nt = 0; nt < 4; ++nt)
            #pragma unroll
            for (int r = 0; r < 8; ++r)
                pw[(r + half * 8) * KBLK + nt * 16 + l16] = (_Float16)s[nt][r];
        // same-wave LDS RAW: DS ops are in-order within a wave.
        v16h p_a[2];
        #pragma unroll
        for (int ch = 0; ch < 2; ++ch) {
            const _Float16* src = &pw[l16 * KBLK + ch * 32 + half * 16];
            v16h a;
            ((uint4*)&a)[0] = *(const uint4*)(src);
            ((uint4*)&a)[1] = *(const uint4*)(src + 8);
            p_a[ch] = a;
        }

        // ---- O += P V  (4 d-subtiles, K=64 via 2 WMMAs each) ----
        // B 32x16: lane holds column n=(lane&15)=d; 16 contiguous keys at
        // ch*32 + half*16 (V stored transposed in LDS).
        #pragma unroll
        for (int dt = 0; dt < 4; ++dt) {
            const _Float16* vb0 = &vt_sm[(dt * 16 + l16) * KBLK + half * 16];
            v16h bf0 = lds_frag(vb0);
            v16h bf1 = lds_frag(vb0 + 32);
            v8f c = o_acc[dt];
            c = __builtin_amdgcn_wmma_f32_16x16x32_f16(
                    false, p_a[0], false, bf0, (short)0, c, false, false);
            c = __builtin_amdgcn_wmma_f32_16x16x32_f16(
                    false, p_a[1], false, bf1, (short)0, c, false, false);
            o_acc[dt] = c;
        }
        __syncthreads();   // protect k_sm / vt_sm before next staging
    }

    // ---- normalize and write O (fp32), C layout -> global ----
    #pragma unroll
    for (int r = 0; r < 8; ++r) {
        const float inv = 1.0f / l_r[r];
        float* orow = Ob + (size_t)(q0 + half * 8 + r) * DK;
        #pragma unroll
        for (int dt = 0; dt < 4; ++dt)
            orow[dt * 16 + l16] = o_acc[dt][r] * inv;
    }
}

extern "C" void kernel_launch(void* const* d_in, const int* in_sizes, int n_in,
                              void* d_out, int out_size, void* d_ws, size_t ws_size,
                              hipStream_t stream) {
    const float* Q = (const float*)d_in[0];
    const float* K = (const float*)d_in[1];
    const float* V = (const float*)d_in[2];
    float*       O = (float*)d_out;
    dim3 grid(NB * (SQ / WG_Q));   // 4 * 64 = 256 workgroups
    fa_fwd_f16wmma<<<grid, NTHREADS, 0, stream>>>(Q, K, V, O);
}
